// RegressionHeadPerceiver_84456236908615
// MI455X (gfx1250) — matrix operand order repaired and checked by reading, not compiled, not run on hardware
//
#include <hip/hip_runtime.h>
#include <hip/hip_bf16.h>

// ---------------------------------------------------------------------------
// RegressionHeadPerceiver for MI455X (gfx1250, wave32, WMMA).
//
// out = mask * (gelu(x@W1 + b1) @ W2 + b2), contiguous slice past prefix ->
// two GEMMs with M=6144 (tokens s in [1024,4096)), K=N=4096. Compute-bound
// (~412 GFLOP vs ~330MB HBM) -> bf16 WMMA with f32 accumulation.
//
// Staging: A tiles via GLOBAL_LOAD_ASYNC_TO_LDS_B128 (ASYNCcnt, no VGPR
// round-trip), B tiles via regs + transpose scatter (layout change needed for
// the WMMA B fragment's K-contiguous per-lane layout). Double-buffered LDS,
// one barrier per K-step.
// ---------------------------------------------------------------------------

typedef __attribute__((ext_vector_type(16))) __bf16 v16bf;
typedef __attribute__((ext_vector_type(8)))  float  v8f;

#define TEXT_H 4096
#define SEQ    4096
#define MROWS  6144        // 2 * (4096 - 1024)
#define KDIM   4096
#define NDIM   4096
#define ROWS_PER_B 3072

#define BM 128
#define BN 128
#define BK 32
#define LDA 40             // padded LDS row stride in bf16 elements (80B, 16B-aligned)

__device__ __forceinline__ unsigned short f32_to_bf16(float f) {
    unsigned int u = __float_as_uint(f);
    u += 0x7FFFu + ((u >> 16) & 1u);   // round-to-nearest-even
    return (unsigned short)(u >> 16);
}

// ---------------- fp32 -> bf16 conversion kernels (workspace staging) ------

__global__ void cvt_w_kernel(const float* __restrict__ src,
                             unsigned short* __restrict__ dst, int n4) {
    int idx = blockIdx.x * blockDim.x + threadIdx.x;
    if (idx >= n4) return;
    float4 f = ((const float4*)src)[idx];
    ushort4 o;
    o.x = f32_to_bf16(f.x); o.y = f32_to_bf16(f.y);
    o.z = f32_to_bf16(f.z); o.w = f32_to_bf16(f.w);
    ((ushort4*)dst)[idx] = o;
}

__global__ void cvt_x_kernel(const float* __restrict__ x,
                             unsigned short* __restrict__ x_bf) {
    const int n4 = MROWS * KDIM / 4;
    int idx = blockIdx.x * blockDim.x + threadIdx.x;
    if (idx >= n4) return;
    int e = idx * 4;
    int m = e >> 12;              // / 4096
    int k = e & 4095;
    int b = m / ROWS_PER_B;
    int s = m - b * ROWS_PER_B + 1024;
    float4 f = *(const float4*)(x + ((size_t)(b * SEQ + s)) * TEXT_H + k);
    ushort4 o;
    o.x = f32_to_bf16(f.x); o.y = f32_to_bf16(f.y);
    o.z = f32_to_bf16(f.z); o.w = f32_to_bf16(f.w);
    *(ushort4*)(x_bf + (size_t)m * KDIM + k) = o;
}

// ---------------- bf16 WMMA GEMM -------------------------------------------
// MODE 0: D = gelu(A@B + bias), stored bf16 (intermediate h)
// MODE 1: D = (A@B + bias) * mask[row], stored fp32 (final output)

template <int MODE>
__global__ __launch_bounds__(256)
void gemm_bf16_wmma(const unsigned short* __restrict__ A,
                    const unsigned short* __restrict__ Bm,
                    const float* __restrict__ bias,
                    const int* __restrict__ mask,
                    unsigned short* __restrict__ out_bf,
                    float* __restrict__ out_f) {
    __shared__ alignas(16) unsigned short sA[2][BM * LDA];  // [buf][row][k]
    __shared__ alignas(16) unsigned short sB[2][BN * LDA];  // [buf][col][k] (transposed)

    const int tid  = threadIdx.x;
    const int lane = tid & 31;
    const int wave = tid >> 5;
    const int wm   = wave & 1;        // wave M slot (2 x 64 rows)
    const int wn   = wave >> 1;       // wave N slot (4 x 32 cols)
    const int hl   = lane >> 4;       // lane half (0/1)
    const int lr   = lane & 15;
    const int m0   = blockIdx.y * BM;
    const int n0   = blockIdx.x * BN;

    // Per-thread staging coordinates (2 x 16B chunks each for A and B tiles).
    int arow[2], ac8[2], kr[2], nc8[2];
#pragma unroll
    for (int it = 0; it < 2; ++it) {
        int idx  = it * 256 + tid;
        arow[it] = idx >> 2;  ac8[it] = idx & 3;    // A: 128 rows x 4 chunks
        kr[it]   = idx >> 4;  nc8[it] = idx & 15;   // B: 32 k-rows x 16 chunks
    }

    v8f acc[4][2];
#pragma unroll
    for (int i = 0; i < 4; ++i)
#pragma unroll
        for (int j = 0; j < 2; ++j)
#pragma unroll
            for (int v = 0; v < 8; ++v) acc[i][j][v] = 0.0f;

    int4 bReg[2];

    // A tile: async DMA global -> LDS, per-lane b128 chunks (ASYNCcnt).
    auto async_loadA = [&](int buf, int kt) {
#pragma unroll
        for (int it = 0; it < 2; ++it) {
            unsigned lds = (unsigned)(size_t)&sA[buf][arow[it] * LDA + ac8[it] * 8];
            const unsigned short* g =
                A + (size_t)(m0 + arow[it]) * KDIM + kt * BK + ac8[it] * 8;
            asm volatile("global_load_async_to_lds_b128 %0, %1, off"
                         :: "v"(lds), "v"(g) : "memory");
        }
    };
    auto wait_async = [&]() { asm volatile("s_wait_asynccnt 0" ::: "memory"); };

    auto load_Bregs = [&](int kt) {
#pragma unroll
        for (int it = 0; it < 2; ++it)
            bReg[it] = *(const int4*)(Bm + (size_t)(kt * BK + kr[it]) * NDIM +
                                      n0 + nc8[it] * 8);
    };
    auto scatter_B = [&](int buf) {
#pragma unroll
        for (int it = 0; it < 2; ++it) {
            union { int4 v; unsigned short u[8]; } t;
            t.v = bReg[it];
#pragma unroll
            for (int e = 0; e < 8; ++e)               // transpose: [k][n] -> [n][k]
                sB[buf][(nc8[it] * 8 + e) * LDA + kr[it]] = t.u[e];
        }
    };

    auto compute = [&](int buf) {
        // B fragment (32x16 bf16): half0 {K0..7,K16..23}, half1 {K8..15,K24..31}
        v16bf bfr[2];
#pragma unroll
        for (int j = 0; j < 2; ++j) {
            union { int4 q[2]; v16bf v; } bf_;
            const unsigned short* pb =
                &sB[buf][(wn * 32 + j * 16 + lr) * LDA + hl * 8];
            bf_.q[0] = *(const int4*)pb;
            bf_.q[1] = *(const int4*)(pb + 16);
            bfr[j] = bf_.v;
        }
#pragma unroll
        for (int i = 0; i < 4; ++i) {
            // A fragment (16x32 bf16): half0 {K0..7,K16..23}, half1 {K8..15,K24..31}
            union { int4 q[2]; v16bf v; } af;
            const unsigned short* pa =
                &sA[buf][(wm * 64 + i * 16 + lr) * LDA + hl * 8];
            af.q[0] = *(const int4*)pa;
            af.q[1] = *(const int4*)(pa + 16);
#pragma unroll
            for (int j = 0; j < 2; ++j) {
                acc[i][j] = __builtin_amdgcn_wmma_f32_16x16x32_bf16(
                    false, af.v, false, bfr[j], (short)0, acc[i][j], false, false);
            }
        }
    };

    const int KT = KDIM / BK;

    // Prologue: stage tile 0 into buffer 0.
    async_loadA(0, 0);
    load_Bregs(0);
    scatter_B(0);
    wait_async();
    __syncthreads();

    int buf = 0;
    for (int kt = 0; kt < KT; ++kt) {
        const bool more = (kt + 1 < KT);
        if (more) {
            async_loadA(buf ^ 1, kt + 1);   // DMA next A tile into other buffer
            load_Bregs(kt + 1);             // next B tile -> regs (hides latency)
        }
        compute(buf);                        // WMMAs on current buffer
        if (more) {
            scatter_B(buf ^ 1);              // publish next B tile
            wait_async();                    // next A tile landed in LDS
            __syncthreads();                 // one barrier per K-step
        }
        buf ^= 1;
    }

    // Epilogue. C/D layout: VGPR v, lane -> row = v + 8*hl, col = lr (per tile).
#pragma unroll
    for (int i = 0; i < 4; ++i) {
#pragma unroll
        for (int j = 0; j < 2; ++j) {
            int col = n0 + wn * 32 + j * 16 + lr;
            float bv = bias[col];
#pragma unroll
            for (int v = 0; v < 8; ++v) {
                int row = m0 + wm * 64 + i * 16 + v + hl * 8;
                float val = acc[i][j][v] + bv;
                if (MODE == 0) {
                    float g = 0.5f * val * (1.0f + erff(val * 0.70710678118654752f));
                    out_bf[(size_t)row * NDIM + col] = f32_to_bf16(g);
                } else {
                    int b = row / ROWS_PER_B;
                    int s = row - b * ROWS_PER_B + 1024;
                    float mv = (mask[b * SEQ + s] != 0) ? 1.0f : 0.0f;
                    out_f[(size_t)row * NDIM + col] = val * mv;
                }
            }
        }
    }
}

// ---------------- launcher --------------------------------------------------

extern "C" void kernel_launch(void* const* d_in, const int* in_sizes, int n_in,
                              void* d_out, int out_size, void* d_ws, size_t ws_size,
                              hipStream_t stream) {
    const float* x    = (const float*)d_in[0];
    const int*   mask = (const int*)d_in[1];
    // d_in[2] = prefix_mask (shape-fixed; unused)
    const float* W1 = (const float*)d_in[3];
    const float* b1 = (const float*)d_in[4];
    const float* W2 = (const float*)d_in[5];
    const float* b2 = (const float*)d_in[6];
    float* out = (float*)d_out;

    char* ws = (char*)d_ws;
    unsigned short* x_bf  = (unsigned short*)(ws);                 // 50,331,648 B
    unsigned short* W1_bf = (unsigned short*)(ws + 50331648);      // 33,554,432 B
    unsigned short* W2_bf = (unsigned short*)(ws + 83886080);      // 33,554,432 B
    unsigned short* h_bf  = (unsigned short*)(ws + 117440512);     // 50,331,648 B

    {   // stage bf16 operands (every launch; deterministic)
        int n4 = MROWS * KDIM / 4;
        cvt_x_kernel<<<(n4 + 255) / 256, 256, 0, stream>>>(x, x_bf);
    }
    {
        int n4 = KDIM * NDIM / 4;
        cvt_w_kernel<<<(n4 + 255) / 256, 256, 0, stream>>>(W1, W1_bf, n4);
        cvt_w_kernel<<<(n4 + 255) / 256, 256, 0, stream>>>(W2, W2_bf, n4);
    }

    dim3 grid(NDIM / BN, MROWS / BM);   // (32, 48)
    dim3 block(256);
    // GEMM1: h = gelu(x@W1 + b1), bf16 out
    gemm_bf16_wmma<0><<<grid, block, 0, stream>>>(x_bf, W1_bf, b1, nullptr, h_bf, nullptr);
    // GEMM2: out = (h@W2 + b2) * mask, fp32 out (already in sliced layout)
    gemm_bf16_wmma<1><<<grid, block, 0, stream>>>(h_bf, W2_bf, b2, mask, nullptr, out);
}
